// MultiHeadSelfAttention_65970697667152
// MI455X (gfx1250) — compile-verified, matrix operand order checked by reading, not run on hardware
//
#include <hip/hip_runtime.h>
#include <hip/hip_bf16.h>
#include <math.h>

typedef __attribute__((ext_vector_type(16))) _Float16 v16h;
typedef __attribute__((ext_vector_type(8)))  _Float16 v8h;
typedef __attribute__((ext_vector_type(8)))  float    v8f;
typedef __attribute__((ext_vector_type(4)))  float    v4f;

#define BATCH 2
#define CCH   256
#define HGT   48
#define WID   48
#define NTOK  2304   // 48*48
#define NHEAD 8
#define DHEAD 32
#define MT    144    // NTOK/16
#define CT    16     // CCH/16
#define SS    2312   // LDS score row stride (floats); 2312 % 64 != 0 -> no bank conflicts

#define WMMA_F16(A, B, C) \
  __builtin_amdgcn_wmma_f32_16x16x32_f16(false, (A), false, (B), (short)0, (C), false, false)

// Load a 16-half WMMA operand tile for this lane. `p` points at the element
// with K==0 for this lane's logical row/col; elements must be contiguous in K.
// Layout (ISA 7.12.2, 16-bit A/B): element e holds K = e + 8*((e>>3)+hf).
// => chunk0 (e=0..7): K = 8*hf + e, chunk1 (e=8..15): K = 16 + 8*hf + (e-8).
// Both chunks are 16-byte aligned v8h loads -> global_load_b128 / ds_load_b128.
__device__ __forceinline__ v16h load_tile_f16(const _Float16* __restrict__ p, int hf) {
  const v8h* q = (const v8h*)p;
  v8h lo = q[hf];       // K = 8*hf .. 8*hf+7
  v8h hi = q[2 + hf];   // K = 16+8*hf .. 23+8*hf
  return __builtin_shufflevector(lo, hi, 0, 1, 2, 3, 4, 5, 6, 7,
                                 8, 9, 10, 11, 12, 13, 14, 15);
}

// ---------------------------------------------------------------------------
// f32 -> f16 conversion (plain)
// ---------------------------------------------------------------------------
__global__ void f32_to_f16_kernel(const float* __restrict__ src,
                                  _Float16* __restrict__ dst, int n) {
  int i = blockIdx.x * blockDim.x + threadIdx.x;
  if (i < n) dst[i] = (_Float16)src[i];
}

// f32 -> f16 with transpose: dst[n*CCH + k] = src[k*CCH + n]  (256x256)
__global__ void f32_to_f16_T_kernel(const float* __restrict__ src,
                                    _Float16* __restrict__ dst) {
  int i = blockIdx.x * blockDim.x + threadIdx.x;   // i = n*CCH + k
  int k = i % CCH, n = i / CCH;
  dst[i] = (_Float16)src[k * CCH + n];
}

// ---------------------------------------------------------------------------
// xs[b,n,c] = f16( x[b,c,h,w] + pe(c,h,w) )
// ---------------------------------------------------------------------------
__global__ void pe_add_kernel(const float* __restrict__ x, _Float16* __restrict__ xs) {
  int idx = blockIdx.x * blockDim.x + threadIdx.x;
  if (idx >= BATCH * NTOK * CCH) return;
  int ci = idx % CCH;
  int t  = (idx / CCH) % NTOK;
  int bb = idx / (CCH * NTOK);
  int yy = t / WID, xx = t % WID;
  int cj = ci & 127;
  int f  = cj & 63;
  float pos = (ci < 128) ? (float)yy : (float)xx;
  float ang = pos * __expf(-(float)f * 0.14391156101f); // 10000^(-f/64)
  float pe  = (cj < 64) ? __sinf(ang) : __cosf(ang);
  float v = x[((bb * CCH + ci) * HGT + yy) * WID + xx] + pe;
  xs[(bb * NTOK + t) * CCH + ci] = (_Float16)v;
}

// ---------------------------------------------------------------------------
// Fused Q/K/V GEMM: one wave per 16x16 output tile, K=256 in 8 WMMA steps.
// B operands come from pre-transposed f16 weights (contiguous-K reads).
// Q,K stored (b,H,n,d); V stored transposed (b,H,d,n) for phase-3 B reads.
// ---------------------------------------------------------------------------
__global__ __launch_bounds__(256) void qkv_gemm_kernel(
    const _Float16* __restrict__ xs,
    const _Float16* __restrict__ wqT, const _Float16* __restrict__ wkT,
    const _Float16* __restrict__ wvT,
    _Float16* __restrict__ qb, _Float16* __restrict__ kb, _Float16* __restrict__ vT) {
  int gw   = blockIdx.x * (blockDim.x >> 5) + (threadIdx.x >> 5);
  int lane = threadIdx.x & 31;
  int hf   = lane >> 4, mrow = lane & 15;
  int bb   = gw / (MT * CT);
  int rest = gw % (MT * CT);
  int mt = rest / CT, nt = rest % CT;

  int col = nt * 16 + mrow;  // output channel for B operand / store
  const _Float16* arow = xs + (size_t)(bb * NTOK + mt * 16 + mrow) * CCH;
  const _Float16* bqrow = wqT + (size_t)col * CCH;
  const _Float16* bkrow = wkT + (size_t)col * CCH;
  const _Float16* bvrow = wvT + (size_t)col * CCH;

  v8f cq = {}, ck = {}, cv = {};
  for (int kc = 0; kc < 8; ++kc) {
    int kbase = kc * 32;
    v16h a  = load_tile_f16(arow + kbase, hf);
    v16h bq = load_tile_f16(bqrow + kbase, hf);
    v16h bk = load_tile_f16(bkrow + kbase, hf);
    v16h bv = load_tile_f16(bvrow + kbase, hf);
    cq = WMMA_F16(a, bq, cq);
    ck = WMMA_F16(a, bk, ck);
    cv = WMMA_F16(a, bv, cv);
  }
  int head = nt >> 1;                 // 16-col tile lies within one head (d=32)
  int dc   = (nt & 1) * 16 + mrow;
  // Q, K: (b,H,n,d) layout (strided f16 stores, 8 per lane)
#pragma unroll
  for (int j = 0; j < 8; ++j) {
    int tok = mt * 16 + 8 * hf + j;
    size_t o = ((size_t)(bb * NHEAD + head) * NTOK + tok) * DHEAD + dc;
    qb[o] = (_Float16)cq[j];
    kb[o] = (_Float16)ck[j];
  }
  // V: (b,H,d,n) layout -> 8 consecutive tokens per lane -> one b128 store
  v8h vs;
#pragma unroll
  for (int j = 0; j < 8; ++j) vs[j] = (_Float16)cv[j];
  *(v8h*)(vT + ((size_t)(bb * NHEAD + head) * DHEAD + dc) * NTOK + mt * 16 + 8 * hf) = vs;
}

// ---------------------------------------------------------------------------
// Attention: one workgroup (8 waves) per (b, head, 16-query block).
// d = 32 == WMMA K -> one WMMA per 16x16 score tile. Full 16x2304 f32 score
// strip in LDS (CDNA5: 320KB/WGP), softmax in place, then P@V via WMMA.
// ---------------------------------------------------------------------------
__global__ __launch_bounds__(256) void attn_kernel(
    const _Float16* __restrict__ qbuf, const _Float16* __restrict__ kbuf,
    const _Float16* __restrict__ vTbuf, _Float16* __restrict__ obuf) {
  __shared__ float scores[16 * SS];          // 147,968 B
  __shared__ float oacc[16][DHEAD];          //   2,048 B
  int bid = blockIdx.x;
  int mt = bid % MT;
  int hh = (bid / MT) % NHEAD;
  int bb = bid / (MT * NHEAD);
  int tid  = threadIdx.x;
  int lane = tid & 31, wave = tid >> 5;
  int hf = lane >> 4, mrow = lane & 15;

  const _Float16* qh = qbuf  + (size_t)(bb * NHEAD + hh) * NTOK * DHEAD;
  const _Float16* kh = kbuf  + (size_t)(bb * NHEAD + hh) * NTOK * DHEAD;
  const _Float16* vh = vTbuf + (size_t)(bb * NHEAD + hh) * DHEAD * NTOK;

  // Q tile for this query block (16x32), loaded once per wave (2x b128)
  v16h aq = load_tile_f16(qh + (mt * 16 + mrow) * DHEAD, hf);

  // Phase 1: scores = Q K^T / sqrt(d)  (144 key tiles, wave-strided)
  for (int nt = wave; nt < MT; nt += 8) {
    v16h bk = load_tile_f16(kh + (nt * 16 + mrow) * DHEAD, hf); // B[k=dcol][n=key]
    v8f cs = {};
    cs = WMMA_F16(aq, bk, cs);
#pragma unroll
    for (int j = 0; j < 8; ++j)
      scores[(j + 8 * hf) * SS + nt * 16 + mrow] = cs[j] * 0.17677669529663689f;
  }
  for (int i = tid; i < 16 * DHEAD; i += 256) ((float*)oacc)[i] = 0.0f;
  __syncthreads();

  // Phase 2: softmax, 16 threads per row (contiguous 16-lane group -> shfl width 16)
  {
    int row = tid >> 4, sub = tid & 15;
    float mx = -3.4e38f;
    for (int i = sub; i < NTOK; i += 16) mx = fmaxf(mx, scores[row * SS + i]);
#pragma unroll
    for (int off = 8; off >= 1; off >>= 1) mx = fmaxf(mx, __shfl_xor(mx, off, 16));
    float sum = 0.0f;
    for (int i = sub; i < NTOK; i += 16) {
      float p = __expf(scores[row * SS + i] - mx);
      scores[row * SS + i] = p;
      sum += p;
    }
#pragma unroll
    for (int off = 8; off >= 1; off >>= 1) sum += __shfl_xor(sum, off, 16);
    float inv = 1.0f / sum;
    for (int i = sub; i < NTOK; i += 16) scores[row * SS + i] *= inv;
  }
  __syncthreads();

  // Phase 3: O = P @ V   (72 K-chunks of 32 across 8 waves, 2 WMMAs/chunk)
  v8f c0 = {}, c1 = {};
  for (int kc = wave; kc < NTOK / 32; kc += 8) {
    int kbase = kc * 32;
    // A = P chunk (16x32) from LDS, f32 -> f16 (contiguous ds_load_b128s)
    const float* sp = &scores[mrow * SS + kbase];
    v16h ap;
#pragma unroll
    for (int e = 0; e < 8; ++e) ap[e] = (_Float16)sp[8 * hf + e];
#pragma unroll
    for (int e = 0; e < 8; ++e) ap[8 + e] = (_Float16)sp[16 + 8 * hf + e];
    // B = V chunk from transposed V: contiguous-K (2x b128 each)
    v16h b0 = load_tile_f16(vh + (size_t)mrow * NTOK + kbase, hf);
    v16h b1 = load_tile_f16(vh + (size_t)(16 + mrow) * NTOK + kbase, hf);
    c0 = WMMA_F16(ap, b0, c0);
    c1 = WMMA_F16(ap, b1, c1);
  }
#pragma unroll
  for (int j = 0; j < 8; ++j) {
    int r = j + 8 * hf;
    atomicAdd(&oacc[r][mrow], c0[j]);
    atomicAdd(&oacc[r][16 + mrow], c1[j]);
  }
  __syncthreads();

  // write O into (b, n, c) f16 layout for the projection GEMM
  for (int i = tid; i < 16 * DHEAD; i += 256) {
    int r = i >> 5, dc = i & 31;
    obuf[(size_t)(bb * NTOK + mt * 16 + r) * CCH + hh * DHEAD + dc] = (_Float16)oacc[r][dc];
  }
}

// ---------------------------------------------------------------------------
// Projection: y = O @ Wp^T + b, written transposed to (b, c, h, w) f32 output
// B[k][n] = Wp[n][k] -> Wp rows are already contiguous in k, no transpose.
// ---------------------------------------------------------------------------
__global__ __launch_bounds__(256) void proj_kernel(
    const _Float16* __restrict__ ob, const _Float16* __restrict__ wp,
    const float* __restrict__ bias, float* __restrict__ out) {
  int gw   = blockIdx.x * (blockDim.x >> 5) + (threadIdx.x >> 5);
  int lane = threadIdx.x & 31;
  int hf   = lane >> 4, mrow = lane & 15;
  int bb   = gw / (MT * CT);
  int rest = gw % (MT * CT);
  int mt = rest / CT, nt = rest % CT;

  int col = nt * 16 + mrow;
  const _Float16* arow = ob + (size_t)(bb * NTOK + mt * 16 + mrow) * CCH;
  const _Float16* brow = wp + (size_t)col * CCH;
  v8f cacc = {};
  for (int kc = 0; kc < 8; ++kc) {
    int kbase = kc * 32;
    v16h a    = load_tile_f16(arow + kbase, hf);
    v16h bmat = load_tile_f16(brow + kbase, hf);
    cacc = WMMA_F16(a, bmat, cacc);
  }
  float bv = bias[col];
  // 8 consecutive tokens per lane -> two aligned float4 stores
  float* op = out + (size_t)(bb * CCH + col) * NTOK + mt * 16 + 8 * hf;
  v4f s0, s1;
#pragma unroll
  for (int j = 0; j < 4; ++j) { s0[j] = cacc[j] + bv; s1[j] = cacc[4 + j] + bv; }
  *(v4f*)op = s0;
  *(v4f*)(op + 4) = s1;
}

// ---------------------------------------------------------------------------
extern "C" void kernel_launch(void* const* d_in, const int* in_sizes, int n_in,
                              void* d_out, int out_size, void* d_ws, size_t ws_size,
                              hipStream_t stream) {
  const float* x  = (const float*)d_in[0];
  const float* Wq = (const float*)d_in[1];
  const float* Wk = (const float*)d_in[2];
  const float* Wv = (const float*)d_in[3];
  const float* Wp = (const float*)d_in[4];
  const float* bp = (const float*)d_in[5];
  float* out = (float*)d_out;

  // workspace layout (all _Float16)
  _Float16* ws = (_Float16*)d_ws;
  size_t off = 0;
  _Float16* xs  = ws + off; off += (size_t)BATCH * NTOK * CCH;
  _Float16* wqT = ws + off; off += (size_t)CCH * CCH;
  _Float16* wkT = ws + off; off += (size_t)CCH * CCH;
  _Float16* wvT = ws + off; off += (size_t)CCH * CCH;
  _Float16* wph = ws + off; off += (size_t)CCH * CCH;
  _Float16* qb  = ws + off; off += (size_t)BATCH * NHEAD * NTOK * DHEAD;
  _Float16* kb  = ws + off; off += (size_t)BATCH * NHEAD * NTOK * DHEAD;
  _Float16* vT  = ws + off; off += (size_t)BATCH * NHEAD * DHEAD * NTOK;
  _Float16* obf = ws + off; off += (size_t)BATCH * NTOK * CCH;

  const int WN = CCH * CCH;
  f32_to_f16_T_kernel<<<WN / 256, 256, 0, stream>>>(Wq, wqT);
  f32_to_f16_T_kernel<<<WN / 256, 256, 0, stream>>>(Wk, wkT);
  f32_to_f16_T_kernel<<<WN / 256, 256, 0, stream>>>(Wv, wvT);
  f32_to_f16_kernel<<<WN / 256, 256, 0, stream>>>(Wp, wph, WN);

  int tot = BATCH * NTOK * CCH;
  pe_add_kernel<<<(tot + 255) / 256, 256, 0, stream>>>(x, xs);

  int gemm_waves = BATCH * MT * CT;   // 4608 tiles, 8 waves/block
  qkv_gemm_kernel<<<gemm_waves / 8, 256, 0, stream>>>(xs, wqT, wkT, wvT, qb, kb, vT);

  attn_kernel<<<BATCH * NHEAD * MT, 256, 0, stream>>>(qb, kb, vT, obf);

  proj_kernel<<<gemm_waves / 8, 256, 0, stream>>>(obf, wph, bp, out);
}
